// GraphConv_26414048871034
// MI455X (gfx1250) — matrix operand order repaired and checked by reading, not compiled, not run on hardware
//
#include <hip/hip_runtime.h>

typedef __attribute__((ext_vector_type(16))) _Float16 v16h;
typedef __attribute__((ext_vector_type(8)))  float    v8f;

#define NN 50000
#define EE 800000
#define WAVES 8

__device__ __forceinline__ v8f wmma16(v16h a, v16h b, v8f c) {
  // D = A(16x32 f16) x B(32x16 f16) + C(16x16 f32)
  return __builtin_amdgcn_wmma_f32_16x16x32_f16(false, a, false, b, (short)0, c, false, false);
}

// B-matrix lane layout: lane half h, element j -> K = 8*h + j (j<8), 16 + 8*h + (j-8) (j>=8)
__device__ __forceinline__ int kmapB(int j, int half) { return j < 8 ? 8 * half + j : 16 + 8 * half + (j - 8); }

// ---------------- weight packing: W1/W2/Wr -> f16 B-matrix lane layout ----------------
// wpk halfs: [0,16384) W1: 32 blocks (k,s,t) of 512; [16384,24576) W2: 16 blocks (k,t);
//            [24576,25600) Wr: 2 blocks (t)
__global__ void prep_pack(const float* __restrict__ W1, const float* __restrict__ W2,
                          const float* __restrict__ Wr, _Float16* __restrict__ wpk) {
  int tid = blockIdx.x * blockDim.x + threadIdx.x;
  if (tid >= 25600) return;
  int within = tid & 511;
  int lane = within >> 4, j = within & 15;
  int half = lane >> 4, col = lane & 15;
  float v;
  if (tid < 16384) {
    int bi = tid >> 9;               // 0..31
    int k = bi >> 2, s = (bi >> 1) & 1, t = bi & 1;
    int N = col + 16 * t;
    int Kr = s * 32 + kmapB(j, half);
    v = (Kr < 36) ? W1[(k * 36 + Kr) * 32 + N] : 0.0f;   // zero-pad K 36..63
  } else if (tid < 24576) {
    int bi = (tid - 16384) >> 9;     // 0..15
    int k = bi >> 1, t = bi & 1;
    int N = col + 16 * t;
    v = W2[(k * 32 + kmapB(j, half)) * 32 + N];
  } else {
    int t = (tid - 24576) >> 9;      // 0..1
    int N = col + 16 * t;
    v = Wr[kmapB(j, half) * 32 + N];
  }
  wpk[tid] = (_Float16)v;
}

__global__ void zero_init(float* __restrict__ out, float* __restrict__ cnt) {
  int i = blockIdx.x * 256 + threadIdx.x;
  if (i < NN * 32) out[i] = 0.0f;
  if (i < NN)      cnt[i] = 0.0f;
}

// ---------------- edge kernel: one wave = 16 edges ----------------
__global__ __launch_bounds__(256)
void edge_kernel(const float* __restrict__ x, const float* __restrict__ pos,
                 const int* __restrict__ ei, const float* __restrict__ ea,
                 const float* __restrict__ ewt, const _Float16* __restrict__ wpk,
                 const float* __restrict__ b1, const float* __restrict__ b2,
                 float* __restrict__ out, float* __restrict__ cnt) {
  __shared__ __align__(32) _Float16 wlds[24576];          // packed W1+W2 (48KB)
  __shared__ __align__(16) _Float16 hbuf[WAVES][16][48];  // h staging, padded rows
  int tid = threadIdx.x;
  { // cooperative copy of packed weights into LDS (L2-hot source, read once per block)
    const float4* s = (const float4*)wpk;
    float4* d = (float4*)wlds;
    for (int i = tid; i < 3072; i += 256) d[i] = s[i];
  }
  __syncthreads();

  int wv = tid >> 5, lane = tid & 31;
  int half = lane >> 4, row = lane & 15;
  int tile = blockIdx.x * WAVES + wv;
  int e = tile * 16 + row;                 // both lane-halves mirror the same edge

  int srcN = ei[e], tgtN = ei[EE + e];
  float dx = pos[srcN * 2]     - pos[tgtN * 2];
  float dy = pos[srcN * 2 + 1] - pos[tgtN * 2 + 1];
  int idx = (dx > 0.f ? 1 : 0) + (dy > 0.f ? 2 : 0) + ((fabsf(dx) - fabsf(dy)) > 0.f ? 4 : 0);
  float ew = ewt[e];

  // wave-uniform branch mask + packed per-row branch ids (3 bits x 16 rows)
  unsigned mask = 1u << idx;
  unsigned long long pack = (unsigned long long)idx << (3 * row);
  for (int off = 16; off > 0; off >>= 1) {
    mask |= __shfl_xor(mask, off);
    pack |= __shfl_xor(pack, off);
  }

  // A-matrix (ISA layout): half h -> e[0..7]=K h*8.., e[8..15]=K 16+h*8..
  v16h A0;          // K 0..31  : x_j - x_i
  v16h A1m = {};    // K 32..63 : edge_attr in 32..35, rest zero
  {
    const float* xj = x + srcN * 32;
    const float* xi = x + tgtN * 32;
    int k0 = half * 8;
#pragma unroll
    for (int j = 0; j < 8; ++j) A0[j]     = (_Float16)(xj[k0 + j]      - xi[k0 + j]);
#pragma unroll
    for (int j = 0; j < 8; ++j) A0[8 + j] = (_Float16)(xj[16 + k0 + j] - xi[16 + k0 + j]);
    if (half == 0) {
#pragma unroll
      for (int j = 0; j < 4; ++j) A1m[j] = (_Float16)ea[e * 4 + j];
    }
  }

  const v16h* WB = (const v16h*)wlds;   // index unit = 16 halves; block = 32 units
  v8f fs0 = {}, fs1 = {};

  for (int k = 0; k < 8; ++k) {
    if (!(mask & (1u << k))) continue;  // wave-uniform skip of absent branches
    // GEMM1: m[16x36] @ W1[k]  (two K-steps, two N-tiles)
    v8f a0 = {}, a1 = {};
    a0 = wmma16(A0,  WB[((k * 2 + 0) * 2 + 0) * 32 + lane], a0);
    a0 = wmma16(A1m, WB[((k * 2 + 1) * 2 + 0) * 32 + lane], a0);
    a1 = wmma16(A0,  WB[((k * 2 + 0) * 2 + 1) * 32 + lane], a1);
    a1 = wmma16(A1m, WB[((k * 2 + 1) * 2 + 1) * 32 + lane], a1);
    float bb0 = b1[k * 32 + row], bb1 = b1[k * 32 + 16 + row];
#pragma unroll
    for (int r = 0; r < 8; ++r) {       // D layout -> LDS (row-major h tile)
      int M = r + half * 8;
      hbuf[wv][M][row]      = (_Float16)fmaxf(a0[r] + bb0, 0.f);
      hbuf[wv][M][16 + row] = (_Float16)fmaxf(a1[r] + bb1, 0.f);
    }
    __builtin_amdgcn_wave_barrier();    // in-wave DS ordering carries the transpose
    v16h A2;
#pragma unroll
    for (int j = 0; j < 8; ++j) A2[j]     = hbuf[wv][row][half * 8 + j];
#pragma unroll
    for (int j = 0; j < 8; ++j) A2[8 + j] = hbuf[wv][row][16 + half * 8 + j];
    __builtin_amdgcn_wave_barrier();
    // GEMM2: h[16x32] @ W2[k]
    v8f c0 = {}, c1 = {};
    c0 = wmma16(A2, WB[1024 + (k * 2 + 0) * 32 + lane], c0);
    c1 = wmma16(A2, WB[1024 + (k * 2 + 1) * 32 + lane], c1);
    float cb0 = b2[k * 32 + row], cb1 = b2[k * 32 + 16 + row];
#pragma unroll
    for (int r = 0; r < 8; ++r) {       // keep rows whose branch id == k
      int M = r + half * 8;
      if ((int)((pack >> (3 * M)) & 7ull) == k) {
        fs0[r] = fmaxf(c0[r] + cb0, 0.f);
        fs1[r] = fmaxf(c1[r] + cb1, 0.f);
      }
    }
  }

  // scatter-add (mean numerator) into out; counts once per edge
#pragma unroll
  for (int r = 0; r < 8; ++r) {
    int M = r + half * 8;
    int tg  = __shfl(tgtN, M);
    float w = __shfl(ew, M);
    atomicAdd(&out[tg * 32 + row],      w * fs0[r]);
    atomicAdd(&out[tg * 32 + 16 + row], w * fs1[r]);
  }
  if (lane < 16) atomicAdd(&cnt[tgtN], 1.0f);
}

// ---------------- node kernel: mean + x@Wr + br (WMMA skip-connection) ----------------
__global__ __launch_bounds__(256)
void node_kernel(const float* __restrict__ x, const _Float16* __restrict__ wbr,
                 const float* __restrict__ br, const float* __restrict__ cnt,
                 float* __restrict__ out) {
  int tid = threadIdx.x;
  int wv = tid >> 5, lane = tid & 31;
  int half = lane >> 4, row = lane & 15;
  int tile = blockIdx.x * WAVES + wv;
  if (tile >= NN / 16) return;          // wave-uniform
  int n = tile * 16 + row;
  const float* xr = x + n * 32;
  int k0 = half * 8;
  v16h A;
#pragma unroll
  for (int j = 0; j < 8; ++j) A[j]     = (_Float16)xr[k0 + j];
#pragma unroll
  for (int j = 0; j < 8; ++j) A[8 + j] = (_Float16)xr[16 + k0 + j];
  const v16h* WB = (const v16h*)wbr;
  v8f a0 = {}, a1 = {};
  a0 = wmma16(A, WB[0 * 32 + lane], a0);
  a1 = wmma16(A, WB[1 * 32 + lane], a1);
  float br0 = br[row], br1 = br[16 + row];
#pragma unroll
  for (int r = 0; r < 8; ++r) {
    int M = r + half * 8;
    int nn = tile * 16 + M;
    float c = fmaxf(cnt[nn], 1.0f);
    out[nn * 32 + row]      = out[nn * 32 + row]      / c + a0[r] + br0;
    out[nn * 32 + 16 + row] = out[nn * 32 + 16 + row] / c + a1[r] + br1;
  }
}

extern "C" void kernel_launch(void* const* d_in, const int* in_sizes, int n_in,
                              void* d_out, int out_size, void* d_ws, size_t ws_size,
                              hipStream_t stream) {
  const float* x   = (const float*)d_in[0];
  const float* pos = (const float*)d_in[1];
  const int*   ei  = (const int*)d_in[2];
  const float* ea  = (const float*)d_in[3];
  const float* ew  = (const float*)d_in[4];
  const float* W1  = (const float*)d_in[5];
  const float* b1  = (const float*)d_in[6];
  const float* W2  = (const float*)d_in[7];
  const float* b2  = (const float*)d_in[8];
  const float* Wr  = (const float*)d_in[9];
  const float* br  = (const float*)d_in[10];
  float* out = (float*)d_out;
  _Float16* wpk = (_Float16*)d_ws;                      // 25600 halfs = 51200 B
  float* cnt = (float*)((char*)d_ws + 51200);           // N floats

  prep_pack<<<100, 256, 0, stream>>>(W1, W2, Wr, wpk);
  zero_init<<<(NN * 32 + 255) / 256, 256, 0, stream>>>(out, cnt);
  edge_kernel<<<EE / (16 * WAVES), 256, 0, stream>>>(x, pos, ei, ea, ew, wpk, b1, b2, out, cnt);
  node_kernel<<<(NN / 16 + WAVES - 1) / WAVES, 256, 0, stream>>>(x, wpk + 24576, br, cnt, out);
}
